// TPALSTM_7980049236134
// MI455X (gfx1250) — matrix operand order repaired and checked by reading, not compiled, not run on hardware
//
#include <hip/hip_runtime.h>
#include <hip/hip_bf16.h>
#include <math.h>
#include <stdint.h>

// ---------------------------------------------------------------------------
// TPA-LSTM on gfx1250 (MI455X): all GEMMs via v_wmma_f32_16x16x32_f16;
// hot LSTM staging uses global_load_async_to_lds_b128 (ASYNCcnt path).
// ---------------------------------------------------------------------------

typedef __attribute__((ext_vector_type(16))) _Float16 v16h;
typedef __attribute__((ext_vector_type(8)))  float    v8f;

#define BDIM 512
#define TDIM 96
#define HDIM 512
#define SEQSTR (TDIM * HDIM)   // row stride (halves) inside (B,T,H) sequence buffers

// ---- CDNA5 async memory->LDS (ISA 10 / 15.18.3, ASYNCcnt) -----------------
__device__ __forceinline__ void async_ld_b128(uint32_t lds_off,
                                              const _Float16* gptr) {
  // per-lane: LDS[lds_off] = MEM[gptr], 16 bytes, tracked by ASYNCcnt
  asm volatile("global_load_async_to_lds_b128 %0, %1, off"
               :: "v"(lds_off), "v"(gptr)
               : "memory");
}

__device__ __forceinline__ void wait_async0() {
#if __has_builtin(__builtin_amdgcn_s_wait_asynccnt)
  __builtin_amdgcn_s_wait_asynccnt(0);
#else
  asm volatile("s_wait_asynccnt 0x0" ::: "memory");
#endif
}

// ---- WMMA fragment gathers from LDS tiles --------------------------------
// A (16x32 f16): lanes 0-15 = rows M, VGPR p holds K=2p,2p+1 (K=0..15);
// lanes 16-31 same rows with K offset +16 (matches frag_b_lds).
__device__ __forceinline__ v16h frag_a_lds(const _Float16* base, int row0,
                                           int pitch, int kc) {
  const int lane = threadIdx.x & 31;
  const int lo = lane & 15, hi = lane >> 4;
  const _Float16* r = base + (row0 + lo) * pitch + kc;
  v16h a;
#pragma unroll
  for (int p = 0; p < 8; ++p) {
    const int k = 2 * p + 16 * hi;
    a[2 * p]     = r[k];
    a[2 * p + 1] = r[k + 1];
  }
  return a;
}

// B (32x16 f16): lane = column N; lanes 0-15 hold K=0..15 contiguous,
// lanes 16-31 hold K=16..31 (contiguous K order, same as frag_a_lds).
__device__ __forceinline__ v16h frag_b_lds(const _Float16* base, int col0,
                                           int pitch, int kc) {
  const int lane = threadIdx.x & 31;
  const int lo = lane & 15, hi = lane >> 4;
  const _Float16* r = base + (col0 + lo) * pitch + kc + hi * 16;
  v16h b;
#pragma unroll
  for (int i = 0; i < 16; ++i) b[i] = r[i];
  return b;
}

__device__ __forceinline__ v8f wmma16(v16h a, v16h b, v8f c) {
  return __builtin_amdgcn_wmma_f32_16x16x32_f16(false, a, false, b,
                                                (short)0, c, false, false);
}

__device__ __forceinline__ float sigm(float x) {
  return 1.f / (1.f + __expf(-x));
}

// ---------------------------------------------------------------------------
// xc = relu(x @ W_hidden^T + b_hidden) + embed[sid]   -> f16 (B*T, 512)
// ---------------------------------------------------------------------------
__global__ __launch_bounds__(256) void xc_kernel(
    const float* __restrict__ x,         // (49152, 64) f32
    const _Float16* __restrict__ Wh,     // (512, 64) f16
    const float* __restrict__ bh,        // (512)
    const float* __restrict__ embed,     // (26, 512)
    const int* __restrict__ sid,         // (512)
    _Float16* __restrict__ xc)           // (49152, 512) f16
{
  __shared__ __align__(16) _Float16 At[64 * 32];
  __shared__ __align__(16) _Float16 Bt[64 * 32];
  const int tid = threadIdx.x;
  const int wave = tid >> 5;
  const int mi = wave >> 1;            // 0..3
  const int ni = wave & 1;             // 0..1
  const int m0 = blockIdx.x * 64;
  const int n0 = blockIdx.y * 64;
  v8f acc[2] = {};

  for (int kc = 0; kc < 64; kc += 32) {
    __syncthreads();
#pragma unroll
    for (int i = 0; i < 2; ++i) {      // A: 64x32, f32 -> f16
      const int id = i * 256 + tid;    // 0..511, 4 floats each
      const int row = id >> 3, c4 = (id & 7) * 4;
      const float4 v4 = *(const float4*)&x[(long)(m0 + row) * 64 + kc + c4];
      _Float16* d = &At[row * 32 + c4];
      d[0] = (_Float16)v4.x; d[1] = (_Float16)v4.y;
      d[2] = (_Float16)v4.z; d[3] = (_Float16)v4.w;
    }
    {                                  // B: 64x32 f16
      const int row = tid >> 2, kq = (tid & 3) * 8;
      *(float4*)&Bt[row * 32 + kq] =
          *(const float4*)&Wh[(long)(n0 + row) * 64 + kc + kq];
    }
    __syncthreads();
    const v16h a = frag_a_lds(At, mi * 16, 32, 0);
#pragma unroll
    for (int s = 0; s < 2; ++s) {
      const v16h b = frag_b_lds(Bt, ni * 32 + s * 16, 32, 0);
      acc[s] = wmma16(a, b, acc[s]);
    }
  }
  const int lane = tid & 31;
  const int lo = lane & 15, hi = lane >> 4;
#pragma unroll
  for (int s = 0; s < 2; ++s) {
    const int n = n0 + ni * 32 + s * 16 + lo;
#pragma unroll
    for (int r = 0; r < 8; ++r) {
      const int m = m0 + mi * 16 + r + 8 * hi;
      float vv = acc[s][r] + bh[n];
      vv = vv > 0.f ? vv : 0.f;
      vv += embed[(long)sid[m / TDIM] * HDIM + n];
      xc[(long)m * HDIM + n] = (_Float16)vv;
    }
  }
}

// ---------------------------------------------------------------------------
// One LSTM timestep, fused:  g = inp@Wih^T + h@Whh^T + bias ; gate math ;
// writes c (in place), h (f16 into sequence buffer) and h (f32 scratch).
// Block tile: M=32 x J=64 (x4 gates). Wave (mi,ji) owns 16x16 for all gates.
// Staging is memory->LDS direct via global_load_async_to_lds_b128.
// ---------------------------------------------------------------------------
__global__ __launch_bounds__(256) void lstm_step_kernel(
    const _Float16* __restrict__ inp,  long istride,  // (512, H)
    const _Float16* __restrict__ hprev, long hstride, // (512, H)
    const _Float16* __restrict__ Wih,                 // (2048, 512)
    const _Float16* __restrict__ Whh,                 // (2048, 512)
    const float* __restrict__ bias,                   // (2048) = b_ih + b_hh
    float* __restrict__ cstate,                       // (512, 512) in/out
    _Float16* __restrict__ hout16, long ostride,
    float* __restrict__ hout32)                       // (512, 512)
{
  __shared__ __align__(16) _Float16 At[32 * 32];
  __shared__ __align__(16) _Float16 Bt[256 * 32];
  const int tid = threadIdx.x;
  const int wave = tid >> 5;
  const int mi = wave >> 2;            // 0..1
  const int ji = wave & 3;             // 0..3
  const int m0 = blockIdx.x * 32;
  const int j0 = blockIdx.y * 64;

  // per-thread staging map (constant across K chunks)
  const int arow = tid >> 2, akq = (tid & 3) * 8;          // A: tid<128 only
  const uint32_t aLds =
      (uint32_t)(uintptr_t)&At[arow * 32 + akq];
  uint32_t bLds[4];
  int bOfs[4];
#pragma unroll
  for (int i = 0; i < 4; ++i) {        // B: 256 rows (4 gates x 64 j) x 32
    const int id = i * 256 + tid;
    const int row = id >> 2, kq = (id & 3) * 8;
    const int g = row >> 6, nl = row & 63;
    bLds[i] = (uint32_t)(uintptr_t)&Bt[row * 32 + kq];
    bOfs[i] = (g * HDIM + j0 + nl) * HDIM + kq;
  }

  v8f acc[4] = {};

  for (int phase = 0; phase < 2; ++phase) {
    const _Float16* Asrc = phase ? hprev : inp;
    const long astr      = phase ? hstride : istride;
    const _Float16* Wsrc = phase ? Whh : Wih;
    const _Float16* aG   = Asrc + (long)(m0 + arow) * astr + akq;
    for (int kc = 0; kc < HDIM; kc += 32) {
      __syncthreads();
      if (tid < 128)                   // A: 32x32 halves, mem -> LDS direct
        async_ld_b128(aLds, aG + kc);
#pragma unroll
      for (int i = 0; i < 4; ++i)      // B: 16 KB tile, mem -> LDS direct
        async_ld_b128(bLds[i], Wsrc + bOfs[i] + kc);
      if (kc + 32 < HDIM)              // hint next weight chunk toward L2
        __builtin_prefetch(&Wsrc[(long)j0 * HDIM + kc + 32 + (tid << 3)], 0, 1);
      wait_async0();
      __syncthreads();
      const v16h a = frag_a_lds(At, mi * 16, 32, 0);
#pragma unroll
      for (int g = 0; g < 4; ++g) {
        const v16h b = frag_b_lds(Bt, g * 64 + ji * 16, 32, 0);
        acc[g] = wmma16(a, b, acc[g]);
      }
    }
  }

  const int lane = tid & 31;
  const int lo = lane & 15, hi = lane >> 4;
  const int n = j0 + ji * 16 + lo;
  const float bi = bias[n];
  const float bf = bias[HDIM + n];
  const float bg = bias[2 * HDIM + n];
  const float bo = bias[3 * HDIM + n];
#pragma unroll
  for (int r = 0; r < 8; ++r) {
    const int m = m0 + mi * 16 + r + 8 * hi;
    const float gi = sigm(acc[0][r] + bi);
    const float gf = sigm(acc[1][r] + bf);
    const float gg = tanhf(acc[2][r] + bg);
    const float go = sigm(acc[3][r] + bo);
    const long ci = (long)m * HDIM + n;
    const float cn = gf * cstate[ci] + gi * gg;
    const float hn = go * tanhf(cn);
    cstate[ci] = cn;
    hout32[ci] = hn;
    hout16[(long)m * ostride + n] = (_Float16)hn;
  }
}

// ---------------------------------------------------------------------------
// conv[b,f,h] = relu( sum_t relu(hseq[b,t,h]) * cw[f,t] + cb[f] )
// Per-batch WMMA GEMM: (32 x 96) . (96 x 512). cw padded to 96 (col 95 = 0).
// ---------------------------------------------------------------------------
__global__ __launch_bounds__(256) void conv_kernel(
    const _Float16* __restrict__ cw,    // (32, 96) f16, padded
    const _Float16* __restrict__ hseq,  // (512, 96, 512) f16
    const float* __restrict__ cb,       // (32)
    float* __restrict__ conv)           // (512, 32, 512) f32
{
  __shared__ __align__(16) _Float16 At[32 * 96];
  __shared__ __align__(16) _Float16 Bt[64 * 96];
  const int tid = threadIdx.x;
  const int wave = tid >> 5;
  const int mi = wave >> 2;            // 0..1
  const int ni = wave & 3;             // 0..3
  const int b = blockIdx.x;
  const int n0 = blockIdx.y * 64;

  for (int id = tid; id < 384; id += 256) {       // A whole: 32x96 halves
    const int row = id / 12, kq = (id % 12) * 8;
    *(float4*)&At[row * 96 + kq] = *(const float4*)&cw[row * 96 + kq];
  }
#pragma unroll
  for (int i = 0; i < 3; ++i) {        // B: transpose-stage with ReLU
    const int id = i * 256 + tid;      // 0..767
    const int t = id >> 3, nq = (id & 7) * 8;
    const float4 v4 =
        *(const float4*)&hseq[((long)b * TDIM + t) * HDIM + n0 + nq];
    const _Float16* hv = (const _Float16*)&v4;
#pragma unroll
    for (int u = 0; u < 8; ++u) {
      const _Float16 h = hv[u];
      Bt[(nq + u) * 96 + t] = (h > (_Float16)0.f) ? h : (_Float16)0.f;
    }
  }
  __syncthreads();
  v8f acc = {};
#pragma unroll
  for (int kc = 0; kc < 96; kc += 32) {
    const v16h a  = frag_a_lds(At, mi * 16, 96, kc);
    const v16h bb = frag_b_lds(Bt, ni * 16, 96, kc);
    acc = wmma16(a, bb, acc);
  }
  const int lane = tid & 31;
  const int lo = lane & 15, hi = lane >> 4;
  const int n = n0 + ni * 16 + lo;
#pragma unroll
  for (int r = 0; r < 8; ++r) {
    const int f = mi * 16 + r + 8 * hi;
    const float vv = acc[r] + cb[f];
    conv[((long)b * 32 + f) * HDIM + n] = vv > 0.f ? vv : 0.f;
  }
}

// ---------------------------------------------------------------------------
// new_ht = [htt | v] @ W2^T + b2  : K = 544 = 17 chunks; chunk 16 comes from
// v (f32, 32 cols), the rest from htt (f16, inside sequence buffer).
// ---------------------------------------------------------------------------
__global__ __launch_bounds__(256) void newht_kernel(
    const _Float16* __restrict__ htt16, long hstride,
    const float* __restrict__ v,        // (512, 32)
    const _Float16* __restrict__ W2,    // (512, 544) f16
    const float* __restrict__ b2,
    float* __restrict__ out)            // (512, 512) f32
{
  __shared__ __align__(16) _Float16 At[64 * 32];
  __shared__ __align__(16) _Float16 Bt[64 * 32];
  const int tid = threadIdx.x;
  const int wave = tid >> 5;
  const int mi = wave >> 1;
  const int ni = wave & 1;
  const int m0 = blockIdx.x * 64;
  const int n0 = blockIdx.y * 64;
  v8f acc[2] = {};

  for (int kc = 0; kc < 544; kc += 32) {
    __syncthreads();
    if (kc < 512) {
      const int row = tid >> 2, kq = (tid & 3) * 8;
      *(float4*)&At[row * 32 + kq] =
          *(const float4*)&htt16[(long)(m0 + row) * hstride + kc + kq];
    } else {
#pragma unroll
      for (int i = 0; i < 2; ++i) {
        const int id = i * 256 + tid;
        const int row = id >> 3, c4 = (id & 7) * 4;
        const float4 v4 = *(const float4*)&v[(long)(m0 + row) * 32 + c4];
        _Float16* d = &At[row * 32 + c4];
        d[0] = (_Float16)v4.x; d[1] = (_Float16)v4.y;
        d[2] = (_Float16)v4.z; d[3] = (_Float16)v4.w;
      }
    }
    {
      const int row = tid >> 2, kq = (tid & 3) * 8;
      *(float4*)&Bt[row * 32 + kq] =
          *(const float4*)&W2[(long)(n0 + row) * 544 + kc + kq];
    }
    __syncthreads();
    const v16h a = frag_a_lds(At, mi * 16, 32, 0);
#pragma unroll
    for (int s = 0; s < 2; ++s) {
      const v16h b = frag_b_lds(Bt, ni * 32 + s * 16, 32, 0);
      acc[s] = wmma16(a, b, acc[s]);
    }
  }
  const int lane = tid & 31;
  const int lo = lane & 15, hi = lane >> 4;
#pragma unroll
  for (int s = 0; s < 2; ++s) {
    const int n = n0 + ni * 32 + s * 16 + lo;
#pragma unroll
    for (int r = 0; r < 8; ++r) {
      const int m = m0 + mi * 16 + r + 8 * hi;
      out[(long)m * HDIM + n] = acc[s][r] + b2[n];
    }
  }
}

// ---------------------------------------------------------------------------
// Small helper kernels.
// ---------------------------------------------------------------------------
__global__ void cvt_f16_kernel(const float* __restrict__ s,
                               _Float16* __restrict__ d, int n) {
  const int i = blockIdx.x * blockDim.x + threadIdx.x;
  if (i < n) d[i] = (_Float16)s[i];
}

__global__ void bias_kernel(const float* __restrict__ a,
                            const float* __restrict__ b,
                            float* __restrict__ o, int n) {
  const int i = blockIdx.x * blockDim.x + threadIdx.x;
  if (i < n) o[i] = a[i] + b[i];
}

__global__ void convw_kernel(const float* __restrict__ cw,   // (32, 95)
                             _Float16* __restrict__ o) {     // (32, 96)
  const int i = blockIdx.x * blockDim.x + threadIdx.x;
  if (i < 32 * 96) {
    const int f = i / 96, t = i % 96;
    o[i] = (t < 95) ? (_Float16)cw[f * 95 + t] : (_Float16)0.f;
  }
}

__global__ void w1_kernel(const float* __restrict__ htt,   // (512, 512)
                          const float* __restrict__ W1,    // (32, 512)
                          const float* __restrict__ b1,    // (32)
                          float* __restrict__ wbuf) {      // (512, 32)
  const int i = blockIdx.x * 256 + threadIdx.x;            // 16384
  const int b = i >> 5, f = i & 31;
  float s = b1[f];
  const float* hr = htt + (long)b * HDIM;
  const float* wr = W1 + (long)f * HDIM;
  for (int k = 0; k < HDIM; ++k) s += hr[k] * wr[k];
  wbuf[i] = s;
}

// feat[b] is the conv block flattened (raw reshape in the reference).
__global__ void attn_kernel(const float* __restrict__ conv,  // (512, 16384)
                            const float* __restrict__ wbuf,  // (512, 32)
                            float* __restrict__ vout) {      // (512, 32)
  __shared__ float ws[32];
  __shared__ float alpha[512];
  const int b = blockIdx.x, tid = threadIdx.x;
  if (tid < 32) ws[tid] = wbuf[b * 32 + tid];
  __syncthreads();
  const float* cb = conv + (long)b * 16384;
  for (int p = tid; p < 512; p += 256) {
    float s = 0.f;
#pragma unroll 8
    for (int q = 0; q < 32; ++q) s += cb[p * 32 + q] * ws[q];
    alpha[p] = 1.f / (1.f + __expf(-s));
  }
  __syncthreads();
  if (tid < 32) {
    float s = 0.f;
    for (int p = 0; p < 512; ++p) s += alpha[p] * cb[p * 32 + tid];
    vout[b * 32 + tid] = s;
  }
}

__global__ void ypred_kernel(const float* __restrict__ nh,    // (512, 512)
                             const float* __restrict__ Wout,  // (96, 512)
                             const float* __restrict__ bout,  // (96)
                             float* __restrict__ out) {       // (512, 24)
  const int i = blockIdx.x * 256 + threadIdx.x;
  if (i >= 512 * 24) return;
  const int b = i / 24, j = i % 24, row = 72 + j;
  float s = bout[row];
  const float* hr = nh + (long)b * HDIM;
  const float* wr = Wout + (long)row * HDIM;
  for (int k = 0; k < HDIM; ++k) s += hr[k] * wr[k];
  out[i] = s;
}

// ---------------------------------------------------------------------------
// Host launcher. Workspace use ~198 MB.
// ---------------------------------------------------------------------------
extern "C" void kernel_launch(void* const* d_in, const int* in_sizes, int n_in,
                              void* d_out, int out_size, void* d_ws,
                              size_t ws_size, hipStream_t stream) {
  (void)in_sizes; (void)n_in; (void)out_size; (void)ws_size;
  const float* x        = (const float*)d_in[0];
  const int*   sid      = (const int*)d_in[1];
  const float* W_hidden = (const float*)d_in[2];
  const float* b_hidden = (const float*)d_in[3];
  const float* embed    = (const float*)d_in[4];
  const float* W_ih     = (const float*)d_in[5];
  const float* W_hh     = (const float*)d_in[6];
  const float* b_ih     = (const float*)d_in[7];
  const float* b_hh     = (const float*)d_in[8];
  const float* conv_W   = (const float*)d_in[9];
  const float* conv_b   = (const float*)d_in[10];
  const float* W1       = (const float*)d_in[11];
  const float* b1       = (const float*)d_in[12];
  const float* W2       = (const float*)d_in[13];
  const float* b2       = (const float*)d_in[14];
  const float* W_out    = (const float*)d_in[15];
  const float* b_out    = (const float*)d_in[16];
  float* out = (float*)d_out;

  char* ws = (char*)d_ws;
  size_t off = 0;
  auto alloc = [&](size_t bytes) -> void* {
    void* p = ws + off;
    off = (off + bytes + 255) & ~(size_t)255;
    return p;
  };
  _Float16* Wh16   = (_Float16*)alloc((size_t)512 * 64 * 2);
  _Float16* Wih16  = (_Float16*)alloc((size_t)2 * 2048 * 512 * 2);
  _Float16* Whh16  = (_Float16*)alloc((size_t)2 * 2048 * 512 * 2);
  float*    biasb  = (float*)   alloc((size_t)2 * 2048 * 4);
  _Float16* cw16   = (_Float16*)alloc((size_t)32 * 96 * 2);
  _Float16* W2_16  = (_Float16*)alloc((size_t)512 * 544 * 2);
  _Float16* xc16   = (_Float16*)alloc((size_t)BDIM * TDIM * HDIM * 2);
  _Float16* h0seq  = (_Float16*)alloc((size_t)BDIM * TDIM * HDIM * 2);
  _Float16* h1seq  = (_Float16*)alloc((size_t)BDIM * TDIM * HDIM * 2);
  _Float16* hzero  = (_Float16*)alloc((size_t)BDIM * HDIM * 2);
  float*    cstate = (float*)   alloc((size_t)BDIM * HDIM * 4);
  float*    h32    = (float*)   alloc((size_t)BDIM * HDIM * 4);
  float*    convb  = (float*)   alloc((size_t)BDIM * 32 * HDIM * 4);
  float*    wbuf   = (float*)   alloc((size_t)BDIM * 32 * 4);
  float*    vbuf   = (float*)   alloc((size_t)BDIM * 32 * 4);
  float*    newht  = (float*)   alloc((size_t)BDIM * HDIM * 4);

  // --- weight prep ---
  cvt_f16_kernel<<<(32768 + 255) / 256, 256, 0, stream>>>(W_hidden, Wh16, 32768);
  cvt_f16_kernel<<<(2097152 + 255) / 256, 256, 0, stream>>>(W_ih, Wih16, 2097152);
  cvt_f16_kernel<<<(2097152 + 255) / 256, 256, 0, stream>>>(W_hh, Whh16, 2097152);
  cvt_f16_kernel<<<(278528 + 255) / 256, 256, 0, stream>>>(W2, W2_16, 278528);
  bias_kernel<<<16, 256, 0, stream>>>(b_ih, b_hh, biasb, 4096);
  convw_kernel<<<12, 256, 0, stream>>>(conv_W, cw16);

  hipMemsetAsync(hzero, 0, (size_t)BDIM * HDIM * 2, stream);
  hipMemsetAsync(cstate, 0, (size_t)BDIM * HDIM * 4, stream);

  // --- input projection + embedding ---
  xc_kernel<<<dim3(BDIM * TDIM / 64, HDIM / 64), 256, 0, stream>>>(
      x, Wh16, b_hidden, embed, sid, xc16);

  // --- LSTM layer 0 (sequential over T) ---
  for (int t = 0; t < TDIM; ++t) {
    const _Float16* inp = xc16 + (long)t * HDIM;
    const _Float16* hp = t ? h0seq + (long)(t - 1) * HDIM : hzero;
    const long hstr = t ? (long)SEQSTR : (long)HDIM;
    lstm_step_kernel<<<dim3(BDIM / 32, HDIM / 64), 256, 0, stream>>>(
        inp, (long)SEQSTR, hp, hstr, Wih16, Whh16, biasb, cstate,
        h0seq + (long)t * HDIM, (long)SEQSTR, h32);
  }

  // --- LSTM layer 1 ---
  hipMemsetAsync(cstate, 0, (size_t)BDIM * HDIM * 4, stream);
  for (int t = 0; t < TDIM; ++t) {
    const _Float16* inp = h0seq + (long)t * HDIM;
    const _Float16* hp = t ? h1seq + (long)(t - 1) * HDIM : hzero;
    const long hstr = t ? (long)SEQSTR : (long)HDIM;
    lstm_step_kernel<<<dim3(BDIM / 32, HDIM / 64), 256, 0, stream>>>(
        inp, (long)SEQSTR, hp, hstr, Wih16 + (long)2048 * 512,
        Whh16 + (long)2048 * 512, biasb + 2048, cstate,
        h1seq + (long)t * HDIM, (long)SEQSTR, h32);
  }
  // h32 now holds htt (f32); h1seq + 95*HDIM holds htt (f16).

  // --- temporal conv + attention + heads ---
  conv_kernel<<<dim3(BDIM, HDIM / 64), 256, 0, stream>>>(cw16, h1seq, conv_b,
                                                         convb);
  w1_kernel<<<64, 256, 0, stream>>>(h32, W1, b1, wbuf);
  attn_kernel<<<BDIM, 256, 0, stream>>>(convb, wbuf, vbuf);
  newht_kernel<<<dim3(BDIM / 64, HDIM / 64), 256, 0, stream>>>(
      h1seq + (long)95 * HDIM, (long)SEQSTR, vbuf, W2_16, b2, newht);
  ypred_kernel<<<48, 256, 0, stream>>>(newht, W_out, b_out, out);
}
